// UpdateAttn_35974646071568
// MI455X (gfx1250) — matrix operand order repaired and checked by reading, not compiled, not run on hardware
//
#include <hip/hip_runtime.h>

// ---------------- types ----------------
typedef __bf16 bf16_t;
typedef __attribute__((ext_vector_type(16))) __bf16 v16bf;
typedef __attribute__((ext_vector_type(8)))  __bf16 v8bf;
typedef __attribute__((ext_vector_type(8)))  float  v8f;
typedef __attribute__((ext_vector_type(4)))  unsigned int u32x4;
typedef __attribute__((ext_vector_type(8)))  int i32x8;
typedef __attribute__((ext_vector_type(4)))  int i32x4;

// ---------------- problem constants ----------------
constexpr int S  = 2048;
constexpr int B  = 2;
constexpr int D  = 1024;
constexpr int NH = 16;
constexpr int DH = 64;
constexpr int M  = S * B;          // 4096 rows (row = i*B + b)

// ---------------- WMMA helpers ----------------
__device__ inline v8f wmma_bf16(v16bf a, v16bf b, v8f c) {
    return __builtin_amdgcn_wmma_f32_16x16x32_bf16(
        /*neg_a=*/false, a, /*neg_b=*/false, b,
        /*c_mod=*/(short)0, c, /*reuse_a=*/false, /*reuse_b=*/false);
}

// A-operand (16x32, row-major source, leading dim ld):
// lane<16: row=lane,   K = {0..7, 16..23}; lane>=16: row=lane-16, K = {8..15, 24..31}
__device__ inline v16bf load_fragA(const bf16_t* tile, int ld) {
    const int lane = threadIdx.x & 31;
    const int row  = lane & 15;
    const int kb   = (lane & 16) ? 8 : 0;
    const bf16_t* p = tile + (size_t)row * ld + kb;
    v8bf lo = *(const v8bf*)(p);
    v8bf hi = *(const v8bf*)(p + 16);
    v16bf f;
#pragma unroll
    for (int t = 0; t < 8; ++t) { f[t] = lo[t]; f[8 + t] = hi[t]; }
    return f;
}

// B-operand (32x16). Source stored "row per output column": Wrow[n][k].
// lane<16: col n=lane, K = {0..7, 16..23}; lane>=16: col n=lane-16, K = {8..15, 24..31}
__device__ inline v16bf load_fragB(const bf16_t* tile, int ld) {
    const int lane = threadIdx.x & 31;
    const int n    = lane & 15;
    const int kb   = (lane & 16) ? 8 : 0;
    const bf16_t* p = tile + (size_t)n * ld + kb;
    v8bf lo = *(const v8bf*)(p);
    v8bf hi = *(const v8bf*)(p + 16);
    v16bf f;
#pragma unroll
    for (int t = 0; t < 8; ++t) { f[t] = lo[t]; f[8 + t] = hi[t]; }
    return f;
}

// ---------------- Tensor Data Mover: 2D tile global -> LDS ----------------
// Stages a [rows x rowlen] bf16 row-major panel (contiguous) into LDS.
// D# built per CDNA5 ISA ch.8: group0 {count, lds_addr, global_addr, type=2},
// group1 {data_size=2B, tensor dims, tile dims, dim0 stride}.
__device__ inline void tdm_load_panel_bf16(const bf16_t* gsrc, bf16_t* lds_dst,
                                           unsigned rows, unsigned rowlen) {
    const unsigned lds_off = (unsigned)(uintptr_t)lds_dst;     // addr[31:0] = LDS offset
    const unsigned long long ga = (unsigned long long)(uintptr_t)gsrc;
    u32x4 g0;
    g0[0] = 1u;                                               // count=1, user mode
    g0[1] = lds_off;                                          // lds_addr
    g0[2] = (unsigned)(ga & 0xFFFFFFFFu);                     // global_addr[31:0]
    g0[3] = (unsigned)((ga >> 32) & 0x01FFFFFFu) | (2u << 30); // addr[56:32] | type=2
    i32x8 g1;
    g1[0] = 0x00010000;                       // wg_mask=0, data_size=1 (2 bytes)
    g1[1] = (int)(rowlen << 16);              // tensor_dim0[15:0] at bit48
    g1[2] = (int)((rowlen >> 16) | (rows << 16));   // dim0 hi | tensor_dim1 lo
    g1[3] = (int)((rows >> 16) | (rowlen << 16));   // dim1 hi | tile_dim0
    g1[4] = (int)rows;                        // tile_dim1 (tile_dim2 = 0)
    g1[5] = (int)rowlen;                      // tensor_dim0_stride[31:0]
    g1[6] = 0;
    g1[7] = 0;
    const i32x4 z4 = {0, 0, 0, 0};
#if __clang_major__ >= 23
    const i32x8 z8 = {0, 0, 0, 0, 0, 0, 0, 0};
    __builtin_amdgcn_tensor_load_to_lds(g0, g1, z4, z4, z8, 0);
#else
    __builtin_amdgcn_tensor_load_to_lds(g0, g1, z4, z4, 0);
#endif
}

// 16x64 output strip: A frag from LDS panel, 4 B tiles from global, K=1024.
__device__ inline void gemm_strip_k1024(const bf16_t* As /*LDS 16x1024*/,
                                        const bf16_t* Brow /*64 rows x 1024*/,
                                        v8f acc[4]) {
#pragma unroll 2
    for (int k = 0; k < 1024; k += 32) {
        __builtin_prefetch(Brow + k + 256, 0, 1);               // global_prefetch_b8
        __builtin_prefetch(Brow + 32 * 1024 + k + 256, 0, 1);
        const v16bf a  = load_fragA(As + k, 1024);              // ds_load
        const v16bf b0 = load_fragB(Brow + k, 1024);
        const v16bf b1 = load_fragB(Brow + 16 * 1024 + k, 1024);
        const v16bf b2 = load_fragB(Brow + 32 * 1024 + k, 1024);
        const v16bf b3 = load_fragB(Brow + 48 * 1024 + k, 1024);
        acc[0] = wmma_bf16(a, b0, acc[0]);
        acc[1] = wmma_bf16(a, b1, acc[1]);
        acc[2] = wmma_bf16(a, b2, acc[2]);
        acc[3] = wmma_bf16(a, b3, acc[3]);
    }
}

// 16-lane (half-wave) reductions
__device__ inline float half16_max(float v) {
#pragma unroll
    for (int m = 1; m < 16; m <<= 1) v = fmaxf(v, __shfl_xor(v, m, 32));
    return v;
}
__device__ inline float half16_sum(float v) {
#pragma unroll
    for (int m = 1; m < 16; m <<= 1) v += __shfl_xor(v, m, 32);
    return v;
}

// ---------------- kernels ----------------
__global__ __launch_bounds__(256) void convert_f32_bf16(const float* __restrict__ src,
                                                        bf16_t* __restrict__ dst, int n) {
    for (int i = blockIdx.x * blockDim.x + threadIdx.x; i < n; i += gridDim.x * blockDim.x)
        dst[i] = (bf16_t)src[i];
}

// QKV projection: block = one 16-row A panel (TDM->LDS) x 512 cols; wave = 16x64 strip.
// Scatter to Q[b][h][S][DH], K[b][h][S][DH], Vt[b][h][DH][S] (bf16).
__global__ __launch_bounds__(256) void gemm_qkv(const bf16_t* __restrict__ X,
                                                const bf16_t* __restrict__ Wqb,
                                                const bf16_t* __restrict__ Wkvb,
                                                bf16_t* __restrict__ Qo,
                                                bf16_t* __restrict__ Ko,
                                                bf16_t* __restrict__ Vto) {
    __shared__ __align__(16) bf16_t As[16 * 1024];              // 32 KB
    const int wave = threadIdx.x >> 5, lane = threadIdx.x & 31;
    const int mt = blockIdx.y;                                  // 256 m-tiles
    const int strip = blockIdx.x;                               // 6 strips of 512
    if (threadIdx.x < 32) {                                     // wave 0 issues the DMA
        tdm_load_panel_bf16(X + (size_t)(mt * 16) * 1024, As, 16u, 1024u);
        __builtin_amdgcn_s_wait_tensorcnt(0);
    }
    __syncthreads();

    const int e0 = strip * 512 + wave * 64;                     // 64-col wave strip
    const bf16_t* Brow = (e0 < D) ? (Wqb + (size_t)e0 * 1024)
                                  : (Wkvb + (size_t)(e0 - D) * 1024);
    v8f acc[4] = {};
    gemm_strip_k1024(As, Brow, acc);

    const int nL = lane & 15, mb = (lane & 16) ? 8 : 0;
#pragma unroll
    for (int c = 0; c < 4; ++c)
#pragma unroll
        for (int r = 0; r < 8; ++r) {
            const int m = mt * 16 + mb + r;
            const int i = m >> 1, b = m & 1;
            const int e = e0 + c * 16 + nL;
            const bf16_t v = (bf16_t)acc[c][r];
            if (e < D) {                       // Q
                const int h = e >> 6, dh = e & 63;
                Qo[(((size_t)(b * NH + h) * S) + i) * DH + dh] = v;
            } else if (e < 2 * D) {            // K
                const int e2 = e - D, h = e2 >> 6, dh = e2 & 63;
                Ko[(((size_t)(b * NH + h) * S) + i) * DH + dh] = v;
            } else {                           // V transposed: [b][h][dh][S]
                const int e3 = e - 2 * D, h = e3 >> 6, dh = e3 & 63;
                Vto[(((size_t)(b * NH + h) * DH) + dh) * S + i] = v;
            }
        }
}

// Flash attention: one wave = 16 query rows of one (b, head); full softmax over j.
__global__ __launch_bounds__(256) void flash_attn(const bf16_t* __restrict__ Q,
                                                  const bf16_t* __restrict__ K,
                                                  const bf16_t* __restrict__ Vt,
                                                  bf16_t* __restrict__ vecb) {
    __shared__ __align__(16) bf16_t Pbuf[8][16][40];   // padded rows: conflict-free
    const int wave = threadIdx.x >> 5, lane = threadIdx.x & 31;
    const int wid = blockIdx.x * 8 + wave;             // 4096 waves total
    const int it = wid & 127;                          // 128 i-tiles
    const int h  = (wid >> 7) & 15;
    const int b  = wid >> 11;
    const int i0 = it * 16;

    const bf16_t* Qh = Q  + ((size_t)(b * NH + h) * S + i0) * DH;
    const bf16_t* Kh = K  + (size_t)(b * NH + h) * S * DH;
    const bf16_t* Vh = Vt + (size_t)(b * NH + h) * DH * S;

    const v16bf q0 = load_fragA(Qh + 0, DH);
    const v16bf q1 = load_fragA(Qh + 32, DH);

    v8f o[4] = {};
    float rmax[8], rsum[8];
#pragma unroll
    for (int r = 0; r < 8; ++r) { rmax[r] = -3.0e38f; rsum[r] = 0.f; }

    bf16_t (*P)[40] = Pbuf[wave];
    const int nL = lane & 15, mb = (lane & 16) ? 8 : 0;
    const float scale = 0.125f;                        // 1/sqrt(64)

    for (int j0 = 0; j0 < S; j0 += 32) {
        // scores: two 16x16 tiles over dh=64 (2 k-steps each)
        v8f s0 = {}, s1 = {};
        {
            const bf16_t* kt0 = Kh + (size_t)j0 * DH;
            const bf16_t* kt1 = Kh + (size_t)(j0 + 16) * DH;
            s0 = wmma_bf16(q0, load_fragB(kt0 + 0,  DH), s0);
            s0 = wmma_bf16(q1, load_fragB(kt0 + 32, DH), s0);
            s1 = wmma_bf16(q0, load_fragB(kt1 + 0,  DH), s1);
            s1 = wmma_bf16(q1, load_fragB(kt1 + 32, DH), s1);
        }
        // online softmax over the 32 new columns
        float p0[8], p1[8], alpha[8];
#pragma unroll
        for (int r = 0; r < 8; ++r) {
            const float a = s0[r] * scale, c = s1[r] * scale;
            const float tmax = half16_max(fmaxf(a, c));
            const float nmax = fmaxf(rmax[r], tmax);
            alpha[r] = __expf(rmax[r] - nmax);
            rmax[r]  = nmax;
            p0[r] = __expf(a - nmax);
            p1[r] = __expf(c - nmax);
            rsum[r] = rsum[r] * alpha[r] + half16_sum(p0[r] + p1[r]);
        }
#pragma unroll
        for (int c = 0; c < 4; ++c)
#pragma unroll
            for (int r = 0; r < 8; ++r) o[c][r] *= alpha[r];

        // stage P (16x32 bf16) to LDS, then reload in A-operand layout
#pragma unroll
        for (int r = 0; r < 8; ++r) {
            P[mb + r][nL]      = (bf16_t)p0[r];
            P[mb + r][16 + nL] = (bf16_t)p1[r];
        }
        asm volatile("s_wait_dscnt 0" ::: "memory");
        const v16bf pf = load_fragA(&P[0][0], 40);

        // O += P * V : 4 dh-chunks, B-operand from transposed V (contiguous)
#pragma unroll
        for (int c = 0; c < 4; ++c)
            o[c] = wmma_bf16(pf, load_fragB(Vh + (size_t)(c * 16) * S + j0, S), o[c]);
    }

    // normalize and scatter to vec[(i*B+b), h*64+dh] as bf16
    float inv[8];
#pragma unroll
    for (int r = 0; r < 8; ++r) inv[r] = 1.0f / rsum[r];
#pragma unroll
    for (int c = 0; c < 4; ++c)
#pragma unroll
        for (int r = 0; r < 8; ++r) {
            const int i = i0 + mb + r;
            const size_t col = (size_t)h * DH + c * 16 + nL;
            vecb[((size_t)i * B + b) * D + col] = (bf16_t)(o[c][r] * inv[r]);
        }
}

// C[4096,1024] = A x W^T + resid (fp32). Block: TDM-staged 16-row panel x 512 cols.
__global__ __launch_bounds__(256) void gemm_add(const bf16_t* __restrict__ A,
                                                const bf16_t* __restrict__ W,
                                                const float* __restrict__ resid,
                                                float* __restrict__ out) {
    __shared__ __align__(16) bf16_t As[16 * 1024];              // 32 KB
    const int wave = threadIdx.x >> 5, lane = threadIdx.x & 31;
    const int mt = blockIdx.y;                                  // 256 m-tiles
    const int nhalf = blockIdx.x;                               // 2 halves of 512
    if (threadIdx.x < 32) {
        tdm_load_panel_bf16(A + (size_t)(mt * 16) * 1024, As, 16u, 1024u);
        __builtin_amdgcn_s_wait_tensorcnt(0);
    }
    __syncthreads();

    const int n0 = nhalf * 512 + wave * 64;
    v8f acc[4] = {};
    gemm_strip_k1024(As, W + (size_t)n0 * 1024, acc);

    const int nL = lane & 15, mb = (lane & 16) ? 8 : 0;
#pragma unroll
    for (int c = 0; c < 4; ++c)
#pragma unroll
        for (int r = 0; r < 8; ++r) {
            const size_t m = mt * 16 + mb + r;
            const size_t e = n0 + c * 16 + nL;
            out[m * D + e] = acc[c][r] + resid[m * D + e];
        }
}

// block-per-row layernorm utilities
__device__ inline void row_stats(const float* row, float v[4], float& mean, float& rstd,
                                 float* sh) {
    float s = 0.f, ss = 0.f;
#pragma unroll
    for (int t = 0; t < 4; ++t) {
        v[t] = row[threadIdx.x + 256 * t];
        s += v[t]; ss += v[t] * v[t];
    }
#pragma unroll
    for (int off = 16; off >= 1; off >>= 1) { s += __shfl_xor(s, off, 32); ss += __shfl_xor(ss, off, 32); }
    const int wave = threadIdx.x >> 5, lane = threadIdx.x & 31;
    if (lane == 0) { sh[wave] = s; sh[8 + wave] = ss; }
    __syncthreads();
    if (wave == 0) {
        float a = (lane < 8) ? sh[lane] : 0.f;
        float c = (lane < 8) ? sh[8 + lane] : 0.f;
#pragma unroll
        for (int off = 4; off >= 1; off >>= 1) { a += __shfl_xor(a, off, 32); c += __shfl_xor(c, off, 32); }
        if (lane == 0) { sh[0] = a; sh[8] = c; }
    }
    __syncthreads();
    mean = sh[0] * (1.f / 1024.f);
    const float var = sh[8] * (1.f / 1024.f) - mean * mean;
    rstd = rsqrtf(var + 1e-5f);
}

__global__ __launch_bounds__(256) void ln_mid(const float* __restrict__ x,
                                              const float* __restrict__ g,
                                              const float* __restrict__ bt,
                                              float* __restrict__ yf,
                                              bf16_t* __restrict__ yb) {
    __shared__ float sh[16];
    const size_t m = blockIdx.x;
    float v[4], mean, rstd;
    row_stats(x + m * D, v, mean, rstd, sh);
#pragma unroll
    for (int t = 0; t < 4; ++t) {
        const int d = threadIdx.x + 256 * t;
        const float y = (v[t] - mean) * rstd * g[d] + bt[d];
        yf[m * D + d] = y;
        yb[m * D + d] = (bf16_t)y;
    }
}

__global__ __launch_bounds__(256) void ln_final(const float* __restrict__ x,
                                                const float* __restrict__ g,
                                                const float* __restrict__ bt,
                                                float* __restrict__ out) {
    __shared__ float sh[16];
    const size_t m = blockIdx.x;
    const int b = (int)(m & 1);
    float v[4], mean, rstd;
    row_stats(x + m * D, v, mean, rstd, sh);
#pragma unroll
    for (int t = 0; t < 4; ++t) {
        const int d = threadIdx.x + 256 * t;
        const float y = (v[t] - mean) * rstd * g[d] + bt[d];
        // pe(b, d): even d -> sin(b * 10000^{-d/D}); odd d -> cos with same freq
        const int j = d >> 1;
        const float f = __expf(-(float)(2 * j) * (9.210340371976184f / 1024.f));
        const float ang = (float)b * f;
        const float pe = (d & 1) ? __cosf(ang) : __sinf(ang);
        out[m * D + d] = 2.f * y + pe;
    }
}

// ---------------- host launcher ----------------
extern "C" void kernel_launch(void* const* d_in, const int* in_sizes, int n_in,
                              void* d_out, int out_size, void* d_ws, size_t ws_size,
                              hipStream_t stream) {
    const float* dec = (const float*)d_in[0];
    const float* Wq  = (const float*)d_in[1];
    const float* Wkv = (const float*)d_in[2];
    const float* Wo  = (const float*)d_in[3];
    const float* g1  = (const float*)d_in[4];
    const float* b1  = (const float*)d_in[5];
    const float* Wff = (const float*)d_in[6];
    const float* g2  = (const float*)d_in[7];
    const float* b2  = (const float*)d_in[8];
    float* out = (float*)d_out;

    char* ws = (char*)d_ws;
    size_t off = 0;
    auto carve = [&](size_t bytes) -> char* {
        char* p = ws + off;
        off += (bytes + 255) & ~(size_t)255;
        return p;
    };
    bf16_t* Xb    = (bf16_t*)carve((size_t)M * D * 2);
    bf16_t* Wqb   = (bf16_t*)carve((size_t)D * D * 2);
    bf16_t* Wkvb  = (bf16_t*)carve((size_t)2 * D * D * 2);
    bf16_t* Wob   = (bf16_t*)carve((size_t)D * D * 2);
    bf16_t* Wffb  = (bf16_t*)carve((size_t)D * D * 2);
    bf16_t* Qb    = (bf16_t*)carve((size_t)B * NH * S * DH * 2);
    bf16_t* Kb    = (bf16_t*)carve((size_t)B * NH * S * DH * 2);
    bf16_t* Vtb   = (bf16_t*)carve((size_t)B * NH * DH * S * 2);
    bf16_t* vecb  = (bf16_t*)carve((size_t)M * D * 2);
    float*  tmp1  = (float*)carve((size_t)M * D * 4);
    float*  out1f = (float*)carve((size_t)M * D * 4);
    bf16_t* out1b = (bf16_t*)carve((size_t)M * D * 2);
    float*  tmp2  = (float*)carve((size_t)M * D * 4);

    // 1) fp32 -> bf16 conversions
    convert_f32_bf16<<<2048, 256, 0, stream>>>(dec, Xb,   M * D);
    convert_f32_bf16<<<1024, 256, 0, stream>>>(Wq,  Wqb,  D * D);
    convert_f32_bf16<<<2048, 256, 0, stream>>>(Wkv, Wkvb, 2 * D * D);
    convert_f32_bf16<<<1024, 256, 0, stream>>>(Wo,  Wob,  D * D);
    convert_f32_bf16<<<1024, 256, 0, stream>>>(Wff, Wffb, D * D);

    // 2) fused QKV projection -> Q, K, V^T (bf16): 256 m-tiles x 6 strips of 512
    gemm_qkv<<<dim3(6, 256), 256, 0, stream>>>(Xb, Wqb, Wkvb, Qb, Kb, Vtb);

    // 3) flash attention -> vec (bf16, [M, D])
    flash_attn<<<(B * NH * (S / 16)) / 8, 256, 0, stream>>>(Qb, Kb, Vtb, vecb);

    // 4) output projection + residual -> tmp1 (fp32)
    gemm_add<<<dim3(2, 256), 256, 0, stream>>>(vecb, Wob, dec, tmp1);

    // 5) layernorm 1 -> out1 (fp32 + bf16)
    ln_mid<<<M, 256, 0, stream>>>(tmp1, g1, b1, out1f, out1b);

    // 6) FF projection + residual -> tmp2 (fp32)
    gemm_add<<<dim3(2, 256), 256, 0, stream>>>(out1b, Wffb, out1f, tmp2);

    // 7) layernorm 2, fused final 2*y + pe(batch) -> d_out
    ln_final<<<M, 256, 0, stream>>>(tmp2, g2, b2, out);
}
